// SemanticRelationUnit_67731634258421
// MI455X (gfx1250) — matrix operand order repaired and checked by reading, not compiled, not run on hardware
//
#include <hip/hip_runtime.h>
#include <hip/hip_bf16.h>
#include <stdint.h>

#define N_OBJ   384
#define IN_DIM  256
#define CLS_DIM 128
#define HID     512
#define OUT_DIM 256
#define FAN_IN  (2*IN_DIM + CLS_DIM)   // 640
#define PAIRS   (N_OBJ*N_OBJ)          // 147456
#define ROWS_PER_BLK 128
#define NBLK    (PAIRS/ROWS_PER_BLK)   // 1152

typedef __attribute__((ext_vector_type(4)))  float        v4f;
typedef __attribute__((ext_vector_type(8)))  float        v8f;
typedef __attribute__((ext_vector_type(8)))  __bf16       v8bf;
typedef __attribute__((ext_vector_type(16))) __bf16       v16bf;
typedef __attribute__((ext_vector_type(4)))  unsigned int u32x4;
typedef __attribute__((ext_vector_type(8)))  int          i32x8;
typedef __attribute__((ext_vector_type(4)))  int          i32x4;

// ---------------- prep kernels ----------------

// a_out[p][i][n] = obj[i,:] . W1[n, p*256 : p*256+256]  (+ b1 when p==0)
__global__ void prep_aij(const float* __restrict__ obj, const float* __restrict__ W1,
                         const float* __restrict__ b1, float* __restrict__ a_out) {
    int idx = blockIdx.x * blockDim.x + threadIdx.x;        // 2*384*512
    int p = idx / (N_OBJ * HID);
    int r = idx - p * (N_OBJ * HID);
    int i = r / HID, n = r % HID;
    const float* x = obj + (size_t)i * IN_DIM;
    const float* w = W1 + (size_t)n * FAN_IN + p * IN_DIM;
    float acc = (p == 0) ? b1[n] : 0.0f;
    for (int k = 0; k < IN_DIM; k += 4) {
        v4f xv = *(const v4f*)(x + k), wv = *(const v4f*)(w + k);
        acc += xv[0]*wv[0] + xv[1]*wv[1] + xv[2]*wv[2] + xv[3]*wv[3];
    }
    a_out[idx] = acc;
}

__global__ void prep_bn(const float* __restrict__ gamma, const float* __restrict__ beta,
                        const float* __restrict__ rm, const float* __restrict__ rv,
                        float* __restrict__ scale, float* __restrict__ shift) {
    int n = blockIdx.x * blockDim.x + threadIdx.x;
    if (n < HID) {
        float s = gamma[n] * rsqrtf(rv[n] + 1e-5f);
        scale[n] = s;
        shift[n] = beta[n] - rm[n] * s;
    }
}

__global__ void prep_w1c(const float* __restrict__ W1, __bf16* __restrict__ w1c) {
    int idx = blockIdx.x * blockDim.x + threadIdx.x;        // 512*128
    int n = idx / CLS_DIM, k = idx % CLS_DIM;
    w1c[idx] = (__bf16)W1[(size_t)n * FAN_IN + 2*IN_DIM + k];
}

__global__ void prep_w2(const float* __restrict__ W2, __bf16* __restrict__ w2b) {
    int idx = blockIdx.x * blockDim.x + threadIdx.x;        // 256*512 row-major [n][k]
    w2b[idx] = (__bf16)W2[idx];
}

// ---------------- fused main kernel ----------------
// Block c: i = c/3, j in [ (c%3)*128, +128 ).  8 waves x 16 rows each.
// Stage 1: h = cls@W1c^T + a_i + a_j  -> leaky -> BN -> bf16 in LDS.
// Stage 2: features = h@W2^T + b2 (W2 tiles staged to LDS via TDM).

__global__ __launch_bounds__(256)
void fused_mlp(const float* __restrict__ cls,
               const float* __restrict__ a_i, const float* __restrict__ a_j,
               const float* __restrict__ scale, const float* __restrict__ shift,
               const __bf16* __restrict__ W1c, const __bf16* __restrict__ W2b,
               const float* __restrict__ b2, float* __restrict__ out_feat) {
    extern __shared__ char smem[];
    __bf16* Hb = (__bf16*)smem;                           // 8 waves * 16 * 512 bf16 = 128 KB
    __bf16* Ws = (__bf16*)(smem + 8 * 16 * HID * 2);      // 16 * 512 bf16 = 16 KB
    const unsigned WS_LDS_OFF = 8u * 16u * HID * 2u;      // byte offset of Ws in LDS

    const int tid = threadIdx.x, wave = tid >> 5, lane = tid & 31;
    const int c = blockIdx.x;
    const int i = c / 3, jbase = (c % 3) * ROWS_PER_BLK;
    const int ml = lane & 15, hi = lane >> 4;

    // ---- A fragments of cls tile (16 rows x 128 K) in bf16 ----
    const int jA = jbase + wave * 16 + ml;
    const float* clsrow = cls + ((size_t)i * N_OBJ + jA) * CLS_DIM;
    v16bf A1[4];
#pragma unroll
    for (int s = 0; s < 4; s++) {
        int k0 = s * 32 + hi * 8;                 // runs per ISA A-layout
        int k1 = k0 + 16;
        v4f q0 = *(const v4f*)(clsrow + k0), q1 = *(const v4f*)(clsrow + k0 + 4);
        v4f q2 = *(const v4f*)(clsrow + k1), q3 = *(const v4f*)(clsrow + k1 + 4);
        v8f f0 = __builtin_shufflevector(q0, q1, 0,1,2,3,4,5,6,7);
        v8f f1 = __builtin_shufflevector(q2, q3, 0,1,2,3,4,5,6,7);
        v8bf lo = __builtin_convertvector(f0, v8bf);
        v8bf hp = __builtin_convertvector(f1, v8bf);
        A1[s] = __builtin_shufflevector(lo, hp, 0,1,2,3,4,5,6,7,8,9,10,11,12,13,14,15);
    }

    const float* ai_row = a_i + (size_t)i * HID;
    __bf16* hb = Hb + wave * 16 * HID;

    // ---- Stage 1: 32 n-tiles x 4 k-steps of v_wmma_f32_16x16x32_bf16 ----
    for (int nt = 0; nt < HID / 16; nt++) {
        v8f acc = {};
        int nlane = nt * 16 + ml;
        const __bf16* w1crow = W1c + (size_t)nlane * CLS_DIM;
#pragma unroll
        for (int s = 0; s < 4; s++) {
            v16bf B = __builtin_shufflevector(*(const v8bf*)(w1crow + s * 32 + hi * 8), *(const v8bf*)(w1crow + s * 32 + 16 + hi * 8), 0,1,2,3,4,5,6,7,8,9,10,11,12,13,14,15);
            acc = __builtin_amdgcn_wmma_f32_16x16x32_bf16(false, A1[s], false, B,
                                                          (short)0, acc, false, false);
        }
        float ain = ai_row[nlane], sc = scale[nlane], sh = shift[nlane];
#pragma unroll
        for (int r = 0; r < 8; r++) {
            int mrow = hi * 8 + r;                           // C layout: VGPR r -> M=r(+8)
            int j = jbase + wave * 16 + mrow;
            float v = acc[r] + ain + a_j[(size_t)j * HID + nlane];
            v = (v >= 0.0f) ? v : 0.01f * v;                 // LeakyReLU
            v = v * sc + sh;                                 // folded BatchNorm
            hb[mrow * HID + nlane] = (__bf16)v;
        }
    }
    __syncthreads();

    // ---- Stage 2: features = h @ W2^T + b2 ----
    const __bf16* hbRow = Hb + wave * 16 * HID + ml * HID;
    for (int ot = 0; ot < OUT_DIM / 16; ot++) {
        // stage W2 rows [ot*16, +16) x 512 (16 KB) into LDS
#if __has_builtin(__builtin_amdgcn_tensor_load_to_lds) && __has_builtin(__builtin_amdgcn_s_wait_tensorcnt)
        if (wave == 0) {
            unsigned long long ga =
                (unsigned long long)(uintptr_t)(W2b + (size_t)ot * 16 * HID);
            u32x4 g0;
            g0[0] = 1u;                                      // count=1 valid descriptor
            g0[1] = WS_LDS_OFF;                              // lds_addr
            g0[2] = (unsigned)(ga & 0xFFFFFFFFu);            // global_addr lo
            g0[3] = (unsigned)((ga >> 32) & 0x1FFFFFFu) | 0x80000000u; // hi | type=2
            i32x8 g1;
            g1[0] = 0x00020000;   // data_size = 4B, no multicast
            g1[1] = 0x01000000;   // tensor_dim0 = 256 dwords (low 16 in [31:16])
            g1[2] = 0x01000000;   // tensor_dim1 = 256 rows
            g1[3] = 0x01000000;   // tile_dim0   = 256 dwords
            g1[4] = 16;           // tile_dim1   = 16 rows
            g1[5] = 256;          // tensor_dim0_stride = 256 dwords
            g1[6] = 0;
            g1[7] = 0;
            i32x4 z4 = {0, 0, 0, 0};
            i32x8 z8 = {0, 0, 0, 0, 0, 0, 0, 0};
            __builtin_amdgcn_tensor_load_to_lds(g0, g1, z4, z4, z8, 0);
            __builtin_amdgcn_s_wait_tensorcnt(0);
        }
#else
        {   // cooperative fallback: 1024 x 16B
            const uint4* src = (const uint4*)(W2b + (size_t)ot * 16 * HID);
            uint4* dst = (uint4*)Ws;
            for (int t = tid; t < 1024; t += 256) dst[t] = src[t];
        }
#endif
        __syncthreads();

        v8f acc = {};
        const __bf16* wsRow = Ws + ml * HID;
#pragma unroll
        for (int s = 0; s < 16; s++) {
            v8bf lo = *(const v8bf*)(hbRow + s * 32 + hi * 8);
            v8bf hp = *(const v8bf*)(hbRow + s * 32 + 16 + hi * 8);
            v16bf A = __builtin_shufflevector(lo, hp, 0,1,2,3,4,5,6,7,8,9,10,11,12,13,14,15);
            v16bf B = __builtin_shufflevector(*(const v8bf*)(wsRow + s * 32 + hi * 8), *(const v8bf*)(wsRow + s * 32 + 16 + hi * 8), 0,1,2,3,4,5,6,7,8,9,10,11,12,13,14,15);
            acc = __builtin_amdgcn_wmma_f32_16x16x32_bf16(false, A, false, B,
                                                          (short)0, acc, false, false);
        }
        int n = ot * 16 + ml;
        float bb = b2[n];
#pragma unroll
        for (int r = 0; r < 8; r++) {
            int mrow = hi * 8 + r;
            size_t row = (size_t)c * ROWS_PER_BLK + wave * 16 + mrow;
            out_feat[row * OUT_DIM + n] = acc[r] + bb;
        }
        __syncthreads();
    }
}

// ---------------- attention / softmax kernel ----------------
__global__ __launch_bounds__(384)
void attn_kernel(const float* __restrict__ feats, const float* __restrict__ Wa,
                 const float* __restrict__ ba, float* __restrict__ out_enh) {
    __shared__ float sE[N_OBJ];
    __shared__ float sStat[2];
    int i = blockIdx.x, t = threadIdx.x;
    const float* frow = feats + ((size_t)i * N_OBJ + t) * OUT_DIM;
    float acc = ba[0];
    for (int o = 0; o < OUT_DIM; o += 4) {
        v4f f = *(const v4f*)(frow + o), w = *(const v4f*)(Wa + o);
        acc += f[0]*w[0] + f[1]*w[1] + f[2]*w[2] + f[3]*w[3];
    }
    sE[t] = acc;
    __syncthreads();
    if (t == 0) {
        float m = sE[0];
        for (int j = 1; j < N_OBJ; j++) m = fmaxf(m, sE[j]);
        sStat[0] = m;
    }
    __syncthreads();
    float e = __expf(sE[t] - sStat[0]);
    __syncthreads();
    sE[t] = e;
    __syncthreads();
    if (t == 0) {
        float s2 = 0.0f;
        for (int j = 0; j < N_OBJ; j++) s2 += sE[j];
        sStat[1] = s2;
    }
    __syncthreads();
    if (t < OUT_DIM) {
        float inv = 1.0f / sStat[1];
        float a2 = 0.0f;
        const float* base = feats + (size_t)i * N_OBJ * OUT_DIM + t;
        for (int j = 0; j < N_OBJ; j++) a2 += sE[j] * base[(size_t)j * OUT_DIM];
        out_enh[i * OUT_DIM + t] = a2 * inv;
    }
}

// ---------------- launcher ----------------
extern "C" void kernel_launch(void* const* d_in, const int* in_sizes, int n_in,
                              void* d_out, int out_size, void* d_ws, size_t ws_size,
                              hipStream_t stream) {
    const float* obj   = (const float*)d_in[0];
    const float* cls   = (const float*)d_in[1];
    const float* W1    = (const float*)d_in[2];
    const float* b1    = (const float*)d_in[3];
    const float* gamma = (const float*)d_in[4];
    const float* beta  = (const float*)d_in[5];
    const float* rm    = (const float*)d_in[6];
    const float* rv    = (const float*)d_in[7];
    const float* W2    = (const float*)d_in[8];
    const float* b2    = (const float*)d_in[9];
    const float* Wa    = (const float*)d_in[10];
    const float* ba    = (const float*)d_in[11];

    float* out_feat = (float*)d_out;                               // [147456, 256]
    float* out_enh  = out_feat + (size_t)PAIRS * OUT_DIM;          // [384, 256]
    float* out_cls  = out_enh + (size_t)N_OBJ * OUT_DIM;           // [384,384,128]

    float*  ws_ai    = (float*)d_ws;                               // [2][384,512]
    float*  ws_aj    = ws_ai + (size_t)N_OBJ * HID;
    float*  ws_scale = ws_aj + (size_t)N_OBJ * HID;                // [512]
    float*  ws_shift = ws_scale + HID;                             // [512]
    __bf16* ws_w1c   = (__bf16*)(ws_shift + HID);                  // [512,128]
    __bf16* ws_w2    = ws_w1c + (size_t)HID * CLS_DIM;             // [256,512]

    prep_aij<<<(2 * N_OBJ * HID) / 256, 256, 0, stream>>>(obj, W1, b1, ws_ai);
    prep_bn <<<2, 256, 0, stream>>>(gamma, beta, rm, rv, ws_scale, ws_shift);
    prep_w1c<<<(HID * CLS_DIM) / 256, 256, 0, stream>>>(W1, ws_w1c);
    prep_w2 <<<(OUT_DIM * HID) / 256, 256, 0, stream>>>(W2, ws_w2);

    size_t lds_bytes = (size_t)8 * 16 * HID * 2 + (size_t)16 * HID * 2; // 144 KB
    fused_mlp<<<NBLK, 256, lds_bytes, stream>>>(cls, ws_ai, ws_aj, ws_scale, ws_shift,
                                                ws_w1c, ws_w2, b2, out_feat);

    attn_kernel<<<N_OBJ, 384, 0, stream>>>(out_feat, Wa, ba, out_enh);

    (void)hipMemcpyAsync(out_cls, cls, (size_t)PAIRS * CLS_DIM * sizeof(float),
                         hipMemcpyDeviceToDevice, stream);
}
